// NmnistMlp_74680891342881
// MI455X (gfx1250) — matrix operand order repaired and checked by reading, not compiled, not run on hardware
//
#include <hip/hip_runtime.h>

// ---------------------------------------------------------------------------
// SLAYER SRM spiking MLP forward for MI455X (gfx1250, wave32, WMMA).
//   L1: u1 = W1 @ s_in   (WMMA f16xf16->f32, K=2312->2336, M=1000->1024)
//       each wave: 32x64 output tile, 8 named accumulators, unroll-1 K loop
//       so C==D tuples stay fixed across the backedge (no v_mov rotation)
//   PSP: 100-tap causal FIR along T (in-place, LDS row buffer)
//   Spike: sequential refractory recurrence, 16-deep register ring
//   L2: u2 = W2 @ spikes (WMMA, K=1024), PSP, Spike -> out
// ---------------------------------------------------------------------------

typedef __attribute__((ext_vector_type(16))) _Float16 v16h;
typedef __attribute__((ext_vector_type(8)))  _Float16 v8h;
typedef __attribute__((ext_vector_type(8)))  float    v8f;

// problem dims
#define NB    32
#define CIN   2312
#define HID   1000
#define COUT  10
#define TT    300
// padded dims
#define KP1   2336          // CIN -> 73 * 32
#define MP1   1024          // HID -> 32 Mtile-pairs of 32 rows
#define TP    320           // TT  -> 20 * 16 (5 groups of 4 tiles)
#define HIDP  1024          // GEMM2 K pad (== MP1)
#define KSTEPS1 (KP1/32)    // 73
#define KSTEPS2 (HIDP/32)   // 32

// A fragment: lane = m + 16*gh; halves j -> K=16*gh+j (contiguous, matches B).
// Two 16B loads, fused into one v16h.
__device__ __forceinline__ v16h load_afrag(const _Float16* __restrict__ p) {
  v8h lo = *(const v8h*)p;
  v8h hi = *(const v8h*)(p + 8);
  return __builtin_shufflevector(lo, hi, 0,1,2,3,4,5,6,7,8,9,10,11,12,13,14,15);
}

#define WMMA_F16(A, B, C) \
  __builtin_amdgcn_wmma_f32_16x16x32_f16(false, (A), false, (B), (short)0, (C), false, false)

// ---------------------------------------------------------------------- packs
__global__ __launch_bounds__(256) void pack_w1(const float* __restrict__ W1,
                                               _Float16* __restrict__ W1h) {
  int e = blockIdx.x * 256 + threadIdx.x;           // MP1*KP1 elements
  int m = e / KP1, k = e - m * KP1;
  float v = (m < HID && k < CIN) ? W1[m * CIN + k] : 0.0f;
  W1h[e] = (_Float16)v;
}

__global__ __launch_bounds__(256) void pack_w2(const float* __restrict__ W2,
                                               _Float16* __restrict__ W2h) {
  int e = blockIdx.x * 256 + threadIdx.x;           // 16*HIDP elements
  int m = e / HIDP, k = e - m * HIDP;
  float v = (m < COUT && k < HID) ? W2[m * HID + k] : 0.0f;
  W2h[e] = (_Float16)v;
}

// transpose-pack s_in[b][c][t] (f32) -> sT[b][t][c] (f16), zero-padded
__global__ __launch_bounds__(256) void pack_sT(const float* __restrict__ s_in,
                                               _Float16* __restrict__ sT) {
  __shared__ float tile[16][17];
  const int c0 = blockIdx.x * 16, t0 = blockIdx.y * 16, b = blockIdx.z;
  const int tx = threadIdx.x, ty = threadIdx.y;
  int c = c0 + ty, t = t0 + tx;
  tile[ty][tx] = (c < CIN && t < TT) ? s_in[((size_t)b * CIN + c) * TT + t] : 0.0f;
  __syncthreads();
  sT[((size_t)b * TP + (t0 + ty)) * KP1 + (c0 + tx)] = (_Float16)tile[tx][ty];
}

// ------------------------------------------------------------------- GEMM 1
// one wave -> 32(M) x 64(N) tile: 2 A frags x 4 B frags -> 8 WMMA per K-step
__global__ __launch_bounds__(256) void gemm1_wmma(
    const _Float16* __restrict__ W1h,   // [MP1][KP1]
    const _Float16* __restrict__ sT,    // [NB][TP][KP1]
    float* __restrict__ u1)             // [NB][MP1][TP]
{
  const int lane = threadIdx.x & 31;
  const int w    = blockIdx.x * 8 + (threadIdx.x >> 5);  // 0..5119
  const int mtp  = w / 160;                              // 0..31 (pair of Mtiles)
  const int g    = w % 160;
  const int b    = g / 5;
  const int tt0  = (g % 5) * 4;

  const int ml = lane & 15;   // A row within tile / D column N
  const int gh = lane >> 4;   // half-wave selector

  const _Float16* arow0 = W1h + (size_t)(mtp * 32 + ml) * KP1 + 16 * gh;
  const _Float16* arow1 = arow0 + (size_t)16 * KP1;
  const _Float16* bbase = sT + ((size_t)b * TP + tt0 * 16 + ml) * KP1 + 16 * gh;

  v8f c00 = {}, c01 = {}, c02 = {}, c03 = {};
  v8f c10 = {}, c11 = {}, c12 = {}, c13 = {};
#pragma unroll 1
  for (int ks = 0; ks < KSTEPS1; ++ks) {
    const int k0 = ks * 32;
    v16h a0 = load_afrag(arow0 + k0);
    v16h a1 = load_afrag(arow1 + k0);
    v16h b0 = *(const v16h*)(bbase + k0);
    v16h b1 = *(const v16h*)(bbase + (size_t)16 * KP1 + k0);
    v16h b2 = *(const v16h*)(bbase + (size_t)32 * KP1 + k0);
    v16h b3 = *(const v16h*)(bbase + (size_t)48 * KP1 + k0);
    c00 = WMMA_F16(a0, b0, c00);
    c01 = WMMA_F16(a0, b1, c01);
    c02 = WMMA_F16(a0, b2, c02);
    c03 = WMMA_F16(a0, b3, c03);
    c10 = WMMA_F16(a1, b0, c10);
    c11 = WMMA_F16(a1, b1, c11);
    c12 = WMMA_F16(a1, b2, c12);
    c13 = WMMA_F16(a1, b3, c13);
  }

  // D layout: lane = N + 16*gh half; element r -> M = 8*gh + r
  const int o0 = mtp * 32 + 8 * gh;
  v8f cc[2][4] = {{c00, c01, c02, c03}, {c10, c11, c12, c13}};
#pragma unroll
  for (int mi = 0; mi < 2; ++mi) {
#pragma unroll
    for (int j = 0; j < 4; ++j) {
      const int t = (tt0 + j) * 16 + ml;
#pragma unroll
      for (int r = 0; r < 8; ++r)
        u1[((size_t)b * MP1 + o0 + 16 * mi + r) * TP + t] = cc[mi][j][r];
    }
  }
}

// ------------------------------------------------------------------- GEMM 2
__global__ __launch_bounds__(256) void gemm2_wmma(
    const _Float16* __restrict__ W2h,   // [16][HIDP]
    const _Float16* __restrict__ hT,    // [NB][TP][HIDP]
    float* __restrict__ u2)             // [NB][16][TP]
{
  const int lane = threadIdx.x & 31;
  const int w    = blockIdx.x * 8 + (threadIdx.x >> 5);  // 0..639
  const int b    = w / 20;
  const int tt   = w % 20;
  const int ml = lane & 15, gh = lane >> 4;

  const _Float16* arow = W2h + (size_t)ml * HIDP + 16 * gh;
  const int t = tt * 16 + ml;
  const _Float16* bcol = hT + ((size_t)b * TP + t) * HIDP + 16 * gh;

  v8f acc = {};
#pragma unroll 1
  for (int ks = 0; ks < KSTEPS2; ++ks) {
    const int k0 = ks * 32;
    v16h a  = load_afrag(arow + k0);
    v16h bf = *(const v16h*)(bcol + k0);
    acc = WMMA_F16(a, bf, acc);
  }
#pragma unroll
  for (int r = 0; r < 8; ++r)
    u2[((size_t)b * 16 + 8 * gh + r) * TP + t] = acc[r];
}

// ------------------------------------------------------------- PSP (FIR 100)
// in-place: one block per (row of TP timesteps); row staged through LDS
__global__ __launch_bounds__(TP) void psp_fir(float* __restrict__ d) {
  __shared__ float row[TP];
  __shared__ float eps[100];
  const int t = threadIdx.x;
  float* r = d + (size_t)blockIdx.x * TP;
  row[t] = r[t];
  if (t < 100) eps[t] = 0.1f * t * expf(1.0f - 0.1f * t);  // eps(0)=0 (causal)
  __syncthreads();
  const int kmax = t < 99 ? t : 99;
  float acc = 0.0f;
  for (int k = 0; k <= kmax; ++k) acc += eps[k] * row[t - k];
  r[t] = acc;
}

// --------------------------------------------------------- Spike (layer 1)
// one thread per (b, o); 16-deep refractory ring fully unrolled -> registers.
// Rows o in [1000,1024) have u1 == 0 (zero weight rows) -> spikes == 0, which
// is exactly the K-padding GEMM2 needs in hT.
__global__ __launch_bounds__(256) void spike1(const float* __restrict__ y1,
                                              _Float16* __restrict__ hT) {
  const int idx = blockIdx.x * 256 + threadIdx.x;   // NB*MP1 threads
  const int b = idx >> 10;
  const int o = idx & 1023;
  float rk[16], R[16];
#pragma unroll
  for (int j = 0; j < 16; ++j) {
    rk[j] = -20.0f * (float)(j + 1) * expf(-(float)j);  // nu(j+1), theta=10
    R[j]  = 0.0f;
  }
  const float* y = y1 + (size_t)idx * TP;
  for (int t = 0; t < TP; ++t) {
    float u = y[t] + R[0];
    float s = (u >= 10.0f) ? 1.0f : 0.0f;
#pragma unroll
    for (int j = 0; j < 15; ++j) R[j] = R[j + 1];
    R[15] = 0.0f;
    if (s != 0.0f) {
#pragma unroll
      for (int j = 0; j < 16; ++j) R[j] += rk[j];
    }
    hT[(((size_t)b * TP + t) << 10) + o] = (_Float16)s;  // [b][t][o] for GEMM2
  }
}

// --------------------------------------------------------- Spike (layer 2)
__global__ __launch_bounds__(320) void spike2(const float* __restrict__ y2,
                                              float* __restrict__ out) {
  const int idx = threadIdx.x;                      // 320 = NB*COUT
  const int b = idx / COUT;
  const int o = idx - b * COUT;
  float rk[16], R[16];
#pragma unroll
  for (int j = 0; j < 16; ++j) {
    rk[j] = -20.0f * (float)(j + 1) * expf(-(float)j);
    R[j]  = 0.0f;
  }
  const float* y = y2 + ((size_t)b * 16 + o) * TP;
  float* dst = out + ((size_t)b * COUT + o) * TT;
  for (int t = 0; t < TT; ++t) {
    float u = y[t] + R[0];
    float s = (u >= 10.0f) ? 1.0f : 0.0f;
#pragma unroll
    for (int j = 0; j < 15; ++j) R[j] = R[j + 1];
    R[15] = 0.0f;
    if (s != 0.0f) {
#pragma unroll
      for (int j = 0; j < 16; ++j) R[j] += rk[j];
    }
    dst[t] = s;
  }
}

// ------------------------------------------------------------------ launcher
extern "C" void kernel_launch(void* const* d_in, const int* in_sizes, int n_in,
                              void* d_out, int out_size, void* d_ws, size_t ws_size,
                              hipStream_t stream) {
  const float* s_in = (const float*)d_in[0];   // [32][2312][300]
  const float* W1   = (const float*)d_in[1];   // [1000][2312]
  const float* W2   = (const float*)d_in[2];   // [10][1000]
  float* out = (float*)d_out;                  // [32][10][300]

  // workspace carve-up (all sizes 256B-multiples), ~116 MB total
  char* ws = (char*)d_ws;
  constexpr size_t SZ_W1H = (size_t)MP1 * KP1 * 2;       //  4,784,128
  constexpr size_t SZ_ST  = (size_t)NB * TP * KP1 * 2;   // 47,841,280
  constexpr size_t SZ_U1  = (size_t)NB * MP1 * TP * 4;   // 41,943,040
  constexpr size_t SZ_HT  = (size_t)NB * TP * HIDP * 2;  // 20,971,520
  constexpr size_t SZ_W2H = (size_t)16 * HIDP * 2;       //     32,768
  _Float16* W1h = (_Float16*)(ws);
  _Float16* sT  = (_Float16*)(ws + SZ_W1H);
  float*    u1  = (float*)   (ws + SZ_W1H + SZ_ST);
  _Float16* hT  = (_Float16*)(ws + SZ_W1H + SZ_ST + SZ_U1);
  _Float16* W2h = (_Float16*)(ws + SZ_W1H + SZ_ST + SZ_U1 + SZ_HT);
  float*    u2  = (float*)   (ws + SZ_W1H + SZ_ST + SZ_U1 + SZ_HT + SZ_W2H);

  // stage inputs to f16 WMMA layouts
  pack_w1<<<(MP1 * KP1) / 256, 256, 0, stream>>>(W1, W1h);
  pack_sT<<<dim3(KP1 / 16, TP / 16, NB), dim3(16, 16), 0, stream>>>(s_in, sT);

  // layer 1: GEMM (32 Mtile-pairs x 160 Ngroups = 5120 waves) -> PSP -> spike
  gemm1_wmma<<<(32 * 160) / 8, 256, 0, stream>>>(W1h, sT, u1);
  psp_fir<<<NB * MP1, TP, 0, stream>>>(u1);
  spike1<<<(NB * MP1) / 256, 256, 0, stream>>>(u1, hT);

  // layer 2: GEMM (640 waves) -> PSP -> spike -> out
  pack_w2<<<(16 * HIDP) / 256, 256, 0, stream>>>(W2, W2h);
  gemm2_wmma<<<640 / 8, 256, 0, stream>>>(W2h, hT, u2);
  psp_fir<<<NB * 16, TP, 0, stream>>>(u2);
  spike2<<<1, 320, 0, stream>>>(u2, out);
}
